// ElasticBertSelfAttention_74363063763242
// MI455X (gfx1250) — matrix operand order repaired and checked by reading, not compiled, not run on hardware
//
#include <hip/hip_runtime.h>
#include <hip/hip_bf16.h>

// MI455X / gfx1250 implementation of ElasticBert self-attention with
// relative_key_query position bias.
//
// Compute-bound => bf16 WMMA everywhere; flash-style streaming so the
// 256 MB score tensor never touches HBM; Tensor Data Mover (TDM) DMAs the
// 2-D tiles into LDS (tracked by TENSORcnt) so waves spend issue slots on
// v_wmma instead of load/ds_store staging.

typedef __bf16 bf16_t;
typedef __attribute__((ext_vector_type(16))) __bf16 v16bf;
typedef __attribute__((ext_vector_type(8)))  float  v8f;
typedef __attribute__((ext_vector_type(4)))  unsigned v4u;
typedef __attribute__((ext_vector_type(4)))  int      v4i;
typedef __attribute__((ext_vector_type(8)))  int      v8i;

#define S_LEN 1024
#define HDIM  1024
#define NH    16
#define DH    64
#define BATCH 4

static __device__ __forceinline__ bf16_t f2bf(float f) {
  union { float f; unsigned u; } a; a.f = f;
  unsigned r = a.u + 0x7FFFu + ((a.u >> 16) & 1u);   // round-to-nearest-even
  union { unsigned short s; bf16_t b; } o; o.s = (unsigned short)(r >> 16);
  return o.b;
}
static __device__ __forceinline__ float bf2f(bf16_t b) {
  union { unsigned short s; bf16_t b; } i; i.b = b;
  union { float f; unsigned u; } o; o.u = ((unsigned)i.s) << 16;
  return o.f;
}
static __device__ __forceinline__ v8f vzero() {
  v8f z;
#pragma unroll
  for (int i = 0; i < 8; ++i) z[i] = 0.0f;
  return z;
}

static __device__ __forceinline__ v8f wmma_bf(v16bf a, v16bf b, v8f c) {
  // D = A(16x32 bf16) * B(32x16 bf16) + C(16x16 f32)
  return __builtin_amdgcn_wmma_f32_16x16x32_bf16(false, a, false, b,
                                                 (short)0, c, false, false);
}

// ---- Tensor Data Mover: DMA a 2-D bf16 tile (tileW x tileH) from global
// (row stride strideElems) into LDS at byte offset ldsOff. D# layout per
// CDNA5 ISA ch.8: group0 {count=1, lds_addr, global_addr, type=2},
// group1 {data_size=1(2B), tensor dims, tile dims, dim0 stride}.
static __device__ __forceinline__ void tdm_load_2d(unsigned ldsOff, const void* g,
                                                   unsigned tileW, unsigned tileH,
                                                   unsigned strideElems,
                                                   unsigned tensorW, unsigned tensorH) {
  unsigned long long ga = (unsigned long long)g;
  v4u g0;
  g0[0] = 1u;                                           // count=1, user mode
  g0[1] = ldsOff;                                       // lds_addr (bytes)
  g0[2] = (unsigned)ga;                                 // global_addr[31:0]
  g0[3] = (unsigned)(ga >> 32) | (2u << 30);            // addr[56:32] | type=2
  v8i g1;
  g1[0] = (int)(1u << 16);                              // data_size=1 (2 bytes)
  g1[1] = (int)((tensorW & 0xFFFFu) << 16);             // tensor_dim0 lo16
  g1[2] = (int)((tensorW >> 16) | ((tensorH & 0xFFFFu) << 16));
  g1[3] = (int)((tensorH >> 16) | ((tileW & 0xFFFFu) << 16)); // | tile_dim0
  g1[4] = (int)(tileH & 0xFFFFu);                       // tile_dim1, tile_dim2=0
  g1[5] = (int)strideElems;                             // tensor_dim0_stride lo32
  g1[6] = 0;                                            // stride hi | dim1_stride lo
  g1[7] = 0;                                            // dim1_stride hi
  v4i z4 = {0, 0, 0, 0};
#if defined(__clang_major__) && (__clang_major__ >= 23)
  v8i z8 = {0, 0, 0, 0, 0, 0, 0, 0};
  __builtin_amdgcn_tensor_load_to_lds(g0, g1, z4, z4, z8, 0);
#else
  __builtin_amdgcn_tensor_load_to_lds(g0, g1, z4, z4, 0);
#endif
}

// A-matrix 16x32 bf16 fragment from LDS tile (row-major, rows = M):
// lanes 0-15: M=lane, elems 0..7 = K 0..7, 8..15 = K 16..23; hi half K+8.
static __device__ __forceinline__ v16bf load_a(const bf16_t* tile, int stride,
                                               int rowBase, int kOff) {
  const int lane = threadIdx.x & 31;
  const int row  = rowBase + (lane & 15);
  const int kb   = kOff + ((lane & 16) ? 8 : 0);
  const bf16_t* p = tile + row * stride + kb;
  union { v16bf v; uint4 q[2]; } f;
  f.q[0] = *reinterpret_cast<const uint4*>(p);
  f.q[1] = *reinterpret_cast<const uint4*>(p + 16);
  return f.v;
}

// B-matrix 32x16 bf16 fragment; tile stored [N][K] contiguous in K:
// lanes 0-15: N=lane, elems 0..7 = K 0..7, 8..15 = K 16..23; hi half K+8
static __device__ __forceinline__ v16bf load_b(const bf16_t* tile, int stride,
                                               int colBase, int kOff) {
  const int lane = threadIdx.x & 31;
  const int col  = colBase + (lane & 15);
  const int kb   = kOff + ((lane & 16) ? 8 : 0);
  const bf16_t* p = tile + col * stride + kb;
  union { v16bf v; uint4 q[2]; } f;
  f.q[0] = *reinterpret_cast<const uint4*>(p);
  f.q[1] = *reinterpret_cast<const uint4*>(p + 16);
  return f.v;
}

__global__ __launch_bounds__(256) void cvt_kernel(const float* __restrict__ src,
                                                  bf16_t* __restrict__ dst, int n) {
  int i = blockIdx.x * blockDim.x + threadIdx.x;
  const int stride = gridDim.x * blockDim.x;
  for (; i < n; i += stride) dst[i] = f2bf(src[i]);
}

// ---------------- QKV projection: [4096,1024] @ [1024,1024] + b ---------------
// grid (16 n-tiles, 64 m-tiles, 3 {q,k,v}); 128 threads = 4 waves; 64x64 tile.
__global__ __launch_bounds__(128) void qkv_kernel(
    const bf16_t* __restrict__ Xb, const bf16_t* __restrict__ Wall,
    const float* __restrict__ bq, const float* __restrict__ bk,
    const float* __restrict__ bv,
    bf16_t* __restrict__ Qb, bf16_t* __restrict__ Kb, bf16_t* __restrict__ Vb) {
  const int tid = threadIdx.x, lane = tid & 31, wave = tid >> 5;
  const int col = lane & 15, hi = (lane & 16) ? 8 : 0;
  const int n0 = blockIdx.x * 64;
  const int m0 = blockIdx.y * 64;
  const int which = blockIdx.z;
  const bf16_t* W = Wall + (size_t)which * HDIM * HDIM;
  const float* bias = (which == 0) ? bq : (which == 1) ? bk : bv;
  bf16_t* dst = (which == 0) ? Qb : (which == 1) ? Kb : Vb;

  // single LDS allocation => At at byte 0 (TDM lds_addr), Bt at 4096
  __shared__ __align__(16) bf16_t smem[64 * 32 * 2];
  bf16_t* At = smem;          // [m][k] 64x32
  bf16_t* Bt = smem + 2048;   // [n][k] 64x32 (transposed W tile)

  v8f acc[4];
#pragma unroll
  for (int nt = 0; nt < 4; ++nt) acc[nt] = vzero();

  for (int k0 = 0; k0 < HDIM; k0 += 32) {
    __syncthreads();
    if (wave == 0)                                       // A: TDM 64x32 tile
      tdm_load_2d(/*ldsOff=*/0, Xb + (size_t)m0 * HDIM + k0,
                  /*tileW=*/32, /*tileH=*/64, /*stride=*/HDIM,
                  /*tensorW=*/HDIM, /*tensorH=*/4096);
    for (int i = tid; i < 256; i += 128) {               // B: transpose on store
      int k = i >> 3, c0 = (i & 7) << 3;
      union { uint4 q; bf16_t h[8]; } t;
      t.q = *reinterpret_cast<const uint4*>(W + (size_t)(k0 + k) * HDIM + n0 + c0);
#pragma unroll
      for (int j = 0; j < 8; ++j) Bt[(c0 + j) * 32 + k] = t.h[j];
    }
    __builtin_amdgcn_s_wait_tensorcnt((short)0);
    __syncthreads();
    v16bf a = load_a(At, 32, wave * 16, 0);
#pragma unroll
    for (int nt = 0; nt < 4; ++nt)
      acc[nt] = wmma_bf(a, load_b(Bt, 32, nt * 16, 0), acc[nt]);
  }

  const int head = n0 >> 6;   // 64-wide tile lies inside one head
#pragma unroll
  for (int nt = 0; nt < 4; ++nt) {
#pragma unroll
    for (int e = 0; e < 8; ++e) {
      int token = m0 + wave * 16 + e + hi;
      int bb = token >> 10, sidx = token & 1023;
      int d = nt * 16 + col;
      float v = acc[nt][e] + bias[n0 + d];
      dst[(((size_t)bb * NH + head) * S_LEN + sidx) * DH + d] = f2bf(v);
    }
  }
}

// ---------------- fused attention with relative_key_query bias ----------------
// grid (16 l-blocks, 64 b*h); 128 threads = 4 waves; each wave owns 16 q-rows.
__global__ __launch_bounds__(128) void attn_kernel(
    const bf16_t* __restrict__ Qg, const bf16_t* __restrict__ Kg,
    const bf16_t* __restrict__ Vg, const bf16_t* __restrict__ Eg,
    const float* __restrict__ mask, float* __restrict__ out) {
  const int tid = threadIdx.x, lane = tid & 31, wave = tid >> 5;
  const int col = lane & 15, hi = (lane & 16) ? 8 : 0;
  const int l0 = blockIdx.x * 64;
  const int bh = blockIdx.y;
  const int b = bh >> 4, h = bh & 15;

  const bf16_t* Qh = Qg + (size_t)bh * S_LEN * DH;
  const bf16_t* Kh = Kg + (size_t)bh * S_LEN * DH;
  const bf16_t* Vh = Vg + (size_t)bh * S_LEN * DH;

  // single LDS allocation: byte offsets Qt=0 Kt=8192 Vt=16384 Et=24576
  // QEc=40960 KEc=49152 (total 57344 B); Pt aliases the dead E region.
  __shared__ __align__(16) bf16_t smem[4096 * 3 + 8192 + 4096 * 2];
  bf16_t* Qt  = smem;          // [64 l][64 d]
  bf16_t* Kt  = Qt + 4096;     // [64 r][64 d]
  bf16_t* Vt  = Kt + 4096;     // [64 d][64 r]   (transposed)
  bf16_t* Et  = Vt + 4096;     // [128 w][64 d]  distance-embedding window
  bf16_t* Pt  = Et;            // [64 l][64 r]   alias: E dead after bias GEMMs
  bf16_t* QEc = Et + 8192;     // [64 l][64]  QEc[l][63-rl] = q[l].E[l-r+2047]
  bf16_t* KEc = QEc + 4096;    // [64 r][64]  KEc[r][ll]    = k[r].E[l-r+2047]

  if (wave == 0)                                          // Q tile once via TDM
    tdm_load_2d(/*ldsOff=*/0, Qh + (size_t)l0 * DH,
                64, 64, DH, DH, S_LEN);

  v8f O[4];
#pragma unroll
  for (int dt = 0; dt < 4; ++dt) O[dt] = vzero();
  float mrow[8], srow[8];
#pragma unroll
  for (int e = 0; e < 8; ++e) { mrow[e] = -1e30f; srow[e] = 0.0f; }

  for (int r0 = 0; r0 < S_LEN; r0 += 64) {
    __syncthreads();
    const int base = l0 - r0 + 2047 - 63;                 // window start, >= 961
    if (wave == 0) {
      tdm_load_2d(/*Kt*/ 8192, Kh + (size_t)r0 * DH, 64, 64, DH, DH, S_LEN);
      tdm_load_2d(/*Et*/ 24576, Eg + (size_t)base * DH, 64, 128, DH, DH, 4095);
    }
    for (int i = tid; i < 512; i += 128) {                // V tile, transposed
      int r = i >> 3, c0 = (i & 7) << 3;
      union { uint4 q; bf16_t h8[8]; } t;
      t.q = *reinterpret_cast<const uint4*>(Vh + (size_t)(r0 + r) * DH + c0);
#pragma unroll
      for (int j = 0; j < 8; ++j) Vt[(c0 + j) * 64 + r] = t.h8[j];
    }
    if (r0 + 64 < S_LEN)                                  // global_prefetch_b8
      __builtin_prefetch(Vh + (size_t)(r0 + 64) * DH + tid * 32, 0, 0);
    __builtin_amdgcn_s_wait_tensorcnt((short)0);          // wave0 TDM complete
    __syncthreads();

    // ---- scores: Q Kt
    v16bf aq0 = load_a(Qt, 64, wave * 16, 0);
    v16bf aq1 = load_a(Qt, 64, wave * 16, 32);
    v8f S[4];
#pragma unroll
    for (int nt = 0; nt < 4; ++nt) {
      v8f acc = vzero();
      acc = wmma_bf(aq0, load_b(Kt, 64, nt * 16, 0), acc);
      acc = wmma_bf(aq1, load_b(Kt, 64, nt * 16, 32), acc);
      S[nt] = acc;
    }

    // ---- bias GEMMs against the E window, band-scatter into QEc/KEc
    v16bf ak0 = load_a(Kt, 64, wave * 16, 0);
    v16bf ak1 = load_a(Kt, 64, wave * 16, 32);
#pragma unroll
    for (int et = 0; et < 8; ++et) {
      v16bf be0 = load_b(Et, 64, et * 16, 0);
      v16bf be1 = load_b(Et, 64, et * 16, 32);
      v8f qe = vzero(), ke = vzero();
      qe = wmma_bf(aq0, be0, qe); qe = wmma_bf(aq1, be1, qe);
      ke = wmma_bf(ak0, be0, ke); ke = wmma_bf(ak1, be1, ke);
#pragma unroll
      for (int e = 0; e < 8; ++e) {
        int m = e + hi, widx = et * 16 + col;
        int lloc = wave * 16 + m;                 // QE rows = this wave's l rows
        int jq = widx - lloc;                     // j = 63 - rl
        if (jq >= 0 && jq < 64) QEc[lloc * 64 + jq] = f2bf(qe[e]);
        int rk = wave * 16 + m;                   // KE rows = this wave's r rows
        int jk = widx + rk - 63;                  // j = lloc
        if (jk >= 0 && jk < 64) KEc[rk * 64 + jk] = f2bf(ke[e]);
      }
    }
    __syncthreads();

    // ---- combine + scale + mask, online softmax
    float tmax[8];
#pragma unroll
    for (int e = 0; e < 8; ++e) tmax[e] = -1e30f;
#pragma unroll
    for (int nt = 0; nt < 4; ++nt) {
      int rl = nt * 16 + col;
      float mk = mask[(size_t)b * S_LEN + r0 + rl];
#pragma unroll
      for (int e = 0; e < 8; ++e) {
        int lloc = wave * 16 + e + hi;
        float v = (S[nt][e] + bf2f(QEc[lloc * 64 + (63 - rl)]) +
                   bf2f(KEc[rl * 64 + lloc])) * 0.125f + mk;
        S[nt][e] = v;
        tmax[e] = fmaxf(tmax[e], v);
      }
    }
#pragma unroll
    for (int e = 0; e < 8; ++e) {
#pragma unroll
      for (int m2 = 1; m2 <= 8; m2 <<= 1)
        tmax[e] = fmaxf(tmax[e], __shfl_xor(tmax[e], m2, 32));
    }
    float corr[8];
#pragma unroll
    for (int e = 0; e < 8; ++e) {
      float mn = fmaxf(mrow[e], tmax[e]);
      corr[e] = __expf(mrow[e] - mn);
      mrow[e] = mn;
      srow[e] *= corr[e];
    }
#pragma unroll
    for (int dt = 0; dt < 4; ++dt)
#pragma unroll
      for (int e = 0; e < 8; ++e) O[dt][e] *= corr[e];

    float psum[8];
#pragma unroll
    for (int e = 0; e < 8; ++e) psum[e] = 0.0f;
#pragma unroll
    for (int nt = 0; nt < 4; ++nt) {
      int rl = nt * 16 + col;
#pragma unroll
      for (int e = 0; e < 8; ++e) {
        float p = __expf(S[nt][e] - mrow[e]);
        psum[e] += p;
        Pt[(wave * 16 + e + hi) * 64 + rl] = f2bf(p);
      }
    }
#pragma unroll
    for (int e = 0; e < 8; ++e) {
#pragma unroll
      for (int m2 = 1; m2 <= 8; m2 <<= 1) psum[e] += __shfl_xor(psum[e], m2, 32);
      srow[e] += psum[e];
    }

    // ---- ctx += P V (wave-local P rows, LDS ordering within wave)
    v16bf ap0 = load_a(Pt, 64, wave * 16, 0);
    v16bf ap1 = load_a(Pt, 64, wave * 16, 32);
#pragma unroll
    for (int dt = 0; dt < 4; ++dt) {
      O[dt] = wmma_bf(ap0, load_b(Vt, 64, dt * 16, 0), O[dt]);
      O[dt] = wmma_bf(ap1, load_b(Vt, 64, dt * 16, 32), O[dt]);
    }
  }

  // write ctx [B,S,1024] fp32
#pragma unroll
  for (int dt = 0; dt < 4; ++dt) {
#pragma unroll
    for (int e = 0; e < 8; ++e) {
      int lg = l0 + wave * 16 + e + hi;
      int c = h * DH + dt * 16 + col;
      out[((size_t)b * S_LEN + lg) * HDIM + c] = O[dt][e] / srow[e];
    }
  }
}

extern "C" void kernel_launch(void* const* d_in, const int* in_sizes, int n_in,
                              void* d_out, int out_size, void* d_ws, size_t ws_size,
                              hipStream_t stream) {
  (void)in_sizes; (void)n_in; (void)out_size; (void)ws_size;
  const float* hs   = (const float*)d_in[0];
  const float* mask = (const float*)d_in[1];
  const float* Wq   = (const float*)d_in[2];
  const float* bq   = (const float*)d_in[3];
  const float* Wk   = (const float*)d_in[4];
  const float* bk   = (const float*)d_in[5];
  const float* Wv   = (const float*)d_in[6];
  const float* bv   = (const float*)d_in[7];
  const float* de   = (const float*)d_in[8];
  float* out = (float*)d_out;

  // workspace layout (bf16 elements), total ~38.5 MiB
  bf16_t* Xb = (bf16_t*)d_ws;                         // 4096*1024
  bf16_t* Wb = Xb + (size_t)4096 * 1024;              // 3 * 1024*1024
  bf16_t* Eb = Wb + (size_t)3 * 1024 * 1024;          // 4095*64 (padded to 4096)
  bf16_t* Qb = Eb + (size_t)4096 * 64;                // 4*16*1024*64 each
  bf16_t* Kb = Qb + (size_t)BATCH * NH * S_LEN * DH;
  bf16_t* Vb = Kb + (size_t)BATCH * NH * S_LEN * DH;

  cvt_kernel<<<1024, 256, 0, stream>>>(hs, Xb, 4096 * 1024);
  cvt_kernel<<<512, 256, 0, stream>>>(Wq, Wb, 1024 * 1024);
  cvt_kernel<<<512, 256, 0, stream>>>(Wk, Wb + 1024 * 1024, 1024 * 1024);
  cvt_kernel<<<512, 256, 0, stream>>>(Wv, Wb + 2 * 1024 * 1024, 1024 * 1024);
  cvt_kernel<<<256, 256, 0, stream>>>(de, Eb, 4095 * 64);

  qkv_kernel<<<dim3(16, 64, 3), 128, 0, stream>>>(Xb, Wb, bq, bk, bv, Qb, Kb, Vb);
  attn_kernel<<<dim3(16, 64), 128, 0, stream>>>(Qb, Kb, Vb, Eb, mask, out);
}
